// densityPropLSTMCell_44289702756936
// MI455X (gfx1250) — matrix operand order repaired and checked by reading, not compiled, not run on hardware
//
#include <hip/hip_runtime.h>

typedef __attribute__((ext_vector_type(16))) _Float16 v16h;
typedef __attribute__((ext_vector_type(8)))  _Float16 v8h;
typedef __attribute__((ext_vector_type(8)))  float    v8f;

#define UDIM 512
#define DDIM 128
#define BDIM 32

// ---------------- WMMA fragment loaders (16x16x32 f16, wave32) ----------------
// A (16x32, MxK): lanes 0-15 row M=lane, halves K=[0..7]∪[16..23]; lanes 16-31 same
// rows, K=[8..15]∪[24..31].  Two contiguous 16B loads per lane.
__device__ __forceinline__ v16h load_a_frag(const _Float16* __restrict__ A, int lda,
                                            int m_base, int kk, int lane) {
  int m  = m_base + (lane & 15);
  int k0 = kk + ((lane >> 4) << 3);
  const _Float16* p = A + (size_t)m * lda + k0;
  v8h lo = *(const v8h*)(p);
  v8h hi = *(const v8h*)(p + 16);
  v16h r;
#pragma unroll
  for (int t = 0; t < 8; ++t) { r[t] = lo[t]; r[t + 8] = hi[t]; }
  return r;
}

// B (32x16, KxN) from an N-major (transposed) array Bt[n][k]: lanes 0-15 col
// N=lane with K=[0..7]∪[16..23]; lanes 16-31 same col, K=[8..15]∪[24..31].  Two 16B loads.
__device__ __forceinline__ v16h load_b_frag(const _Float16* __restrict__ Bt, int ldb,
                                            int n_base, int kk, int lane) {
  int n  = n_base + (lane & 15);
  int k0 = kk + ((lane >> 4) << 3);
  const _Float16* p = Bt + (size_t)n * ldb + k0;
  v8h lo = *(const v8h*)(p);
  v8h hi = *(const v8h*)(p + 16);
  v16h r;
#pragma unroll
  for (int t = 0; t < 8; ++t) { r[t] = lo[t]; r[t + 8] = hi[t]; }
  return r;
}

// ---------------- generic f32 -> f16 transpose: dst[z][c][r] = src[z][r][c] ----
__global__ __launch_bounds__(256) void transpose_f32_f16(const float* __restrict__ src,
                                                         _Float16* __restrict__ dst,
                                                         int R, int C) {
  __shared__ float tile[32][33];
  int bz = blockIdx.z;
  int r0 = blockIdx.y * 32, c0 = blockIdx.x * 32;
  int tx = threadIdx.x & 31, ty = threadIdx.x >> 5;  // ty: 0..7
  const float* s = src + (size_t)bz * R * C;
  _Float16* d = dst + (size_t)bz * R * C;
#pragma unroll
  for (int yy = 0; yy < 4; ++yy) {
    int r = r0 + ty + yy * 8;
    tile[ty + yy * 8][tx] = s[(size_t)r * C + c0 + tx];
  }
  __syncthreads();
#pragma unroll
  for (int yy = 0; yy < 4; ++yy) {
    int c = ty + yy * 8;
    d[(size_t)(c0 + c) * R + r0 + tx] = (_Float16)tile[tx][c];
  }
}

// ---------------- gates: pre/mu/grad, reductions, diag terms, c/tanh(c) -------
struct GateParams {
  const float* U[4];
  const float* usig[4];
  const float* W[4];
  const float* wsig[4];
};

__global__ __launch_bounds__(512) void gates_kernel(
    const float* __restrict__ x, const float* __restrict__ h,
    const float* __restrict__ cprev, const float* __restrict__ Sh,
    GateParams P,
    float* __restrict__ mu, float* __restrict__ grad, float* __restrict__ diag,
    float* __restrict__ tcv, float* __restrict__ gtcv,
    float* __restrict__ out_mu, float* __restrict__ out_c) {
  __shared__ float xs[DDIM];
  __shared__ float hs[UDIM];
  __shared__ float red[UDIM];
  int b = blockIdx.x, i = threadIdx.x;
  if (i < DDIM) xs[i] = x[b * DDIM + i];
  float hv = h[b * UDIM + i];
  hs[i] = hv;
  __syncthreads();

  // hh = sum h^2
  red[i] = hv * hv; __syncthreads();
  for (int s = 256; s > 0; s >>= 1) { if (i < s) red[i] += red[i + s]; __syncthreads(); }
  float hh = red[0]; __syncthreads();
  // xx = sum x^2
  red[i] = (i < DDIM) ? xs[i] * xs[i] : 0.f; __syncthreads();
  for (int s = 256; s > 0; s >>= 1) { if (i < s) red[i] += red[i + s]; __syncthreads(); }
  float xx = red[0]; __syncthreads();
  // tr(Sigma_h)
  red[i] = Sh[(size_t)b * UDIM * UDIM + (size_t)i * UDIM + i]; __syncthreads();
  for (int s = 256; s > 0; s >>= 1) { if (i < s) red[i] += red[i + s]; __syncthreads(); }
  float tr = red[0];

  float a0 = 0.f, a1 = 0.f, a2 = 0.f, a3 = 0.f;
  for (int d = 0; d < DDIM; ++d) {
    float xv = xs[d];
    a0 += xv * P.U[0][d * UDIM + i];
    a1 += xv * P.U[1][d * UDIM + i];
    a2 += xv * P.U[2][d * UDIM + i];
    a3 += xv * P.U[3][d * UDIM + i];
  }
  for (int j = 0; j < UDIM; ++j) {
    float hj = hs[j];
    a0 += hj * P.W[0][j * UDIM + i];
    a1 += hj * P.W[1][j * UDIM + i];
    a2 += hj * P.W[2][j * UDIM + i];
    a3 += hj * P.W[3][j * UDIM + i];
  }
  float m0 = 1.f / (1.f + expf(-a0));  // f
  float m1 = 1.f / (1.f + expf(-a1));  // i
  float m2 = 1.f / (1.f + expf(-a2));  // o
  float m3 = tanhf(a3);                // g
  float ms[4]  = {m0, m1, m2, m3};
  float gs[4]  = {m0 * (1.f - m0), m1 * (1.f - m1), m2 * (1.f - m2), 1.f - m3 * m3};
  int bi = b * UDIM + i;
  const int GS = BDIM * UDIM;
#pragma unroll
  for (int g = 0; g < 4; ++g) {
    mu[g * GS + bi]   = ms[g];
    grad[g * GS + bi] = gs[g];
    float spu = log1pf(expf(P.usig[g][i]));
    float spw = log1pf(expf(P.wsig[g][i]));
    diag[g * GS + bi] = xx * spu + (hh + tr) * spw;
  }
  float c = cprev[bi] * m0 + m1 * m3;
  float t = tanhf(c);
  out_c[bi]  = c;
  out_mu[bi] = t * m2;
  tcv[bi]    = t;
  gtcv[bi]   = 1.f - t * t;
}

// ---------------- stage 1: T[b] = W^T * Sigma_h[b] (f16 in / f32 acc / f16 out)
__global__ __launch_bounds__(256) void stage1_kernel(const _Float16* __restrict__ Wt,
                                                     const _Float16* __restrict__ St,
                                                     _Float16* __restrict__ T) {
  const int U = UDIM;
  int lane = threadIdx.x & 31, wave = threadIdx.x >> 5;
  int wm = wave & 1, wn = wave >> 1;              // 2 x 4 wave grid
  int b = blockIdx.z;
  int tm0 = blockIdx.x * 128 + wm * 64;           // 4 M-subtiles
  int tn0 = blockIdx.y * 128 + wn * 32;           // 2 N-subtiles
  const _Float16* Bt = St + (size_t)b * U * U;
  v8f acc[4][2] = {};
  for (int kk = 0; kk < U; kk += 32) {
    v16h a[4], bf[2];
#pragma unroll
    for (int mt = 0; mt < 4; ++mt) a[mt] = load_a_frag(Wt, U, tm0 + mt * 16, kk, lane);
#pragma unroll
    for (int nt = 0; nt < 2; ++nt) bf[nt] = load_b_frag(Bt, U, tn0 + nt * 16, kk, lane);
#pragma unroll
    for (int mt = 0; mt < 4; ++mt)
#pragma unroll
      for (int nt = 0; nt < 2; ++nt)
        acc[mt][nt] = __builtin_amdgcn_wmma_f32_16x16x32_f16(
            false, a[mt], false, bf[nt], (short)0, acc[mt][nt], false, false);
  }
  _Float16* Tb = T + (size_t)b * U * U;
  int ln = lane & 15, hb = lane >> 4;
#pragma unroll
  for (int mt = 0; mt < 4; ++mt)
#pragma unroll
    for (int nt = 0; nt < 2; ++nt)
#pragma unroll
      for (int r = 0; r < 8; ++r) {
        int m = tm0 + mt * 16 + r + hb * 8;
        int n = tn0 + nt * 16 + ln;
        Tb[(size_t)m * U + n] = (_Float16)acc[mt][nt][r];
      }
}

// ---------------- stage 2: S = (T[b]*W + diag) .* (grad grad^T), fused epilogue
// mode 0: out = H(Sin,S,m1,m2)    (gate f: Sigma_out_c base)
// mode 1: out = S                 (gates i,o: stash into Sigma_out region)
// mode 2: out += H(Sin,S,m1,m2)   (gate g: accumulate into Sigma_out_c)
__global__ __launch_bounds__(256) void stage2_kernel(
    const _Float16* __restrict__ Tm, const _Float16* __restrict__ Wt,
    const float* __restrict__ grad, const float* __restrict__ diag,
    const float* __restrict__ Sin, const float* __restrict__ m1v,
    const float* __restrict__ m2v, float* __restrict__ outp, int mode) {
  const int U = UDIM;
  int lane = threadIdx.x & 31, wave = threadIdx.x >> 5;
  int wm = wave & 1, wn = wave >> 1;
  int b = blockIdx.z;
  int tm0 = blockIdx.x * 128 + wm * 64;
  int tn0 = blockIdx.y * 128 + wn * 32;
  const _Float16* A = Tm + (size_t)b * U * U;
  v8f acc[4][2] = {};
  for (int kk = 0; kk < U; kk += 32) {
    v16h a[4], bf[2];
#pragma unroll
    for (int mt = 0; mt < 4; ++mt) a[mt] = load_a_frag(A, U, tm0 + mt * 16, kk, lane);
#pragma unroll
    for (int nt = 0; nt < 2; ++nt) bf[nt] = load_b_frag(Wt, U, tn0 + nt * 16, kk, lane);
#pragma unroll
    for (int mt = 0; mt < 4; ++mt)
#pragma unroll
      for (int nt = 0; nt < 2; ++nt)
        acc[mt][nt] = __builtin_amdgcn_wmma_f32_16x16x32_f16(
            false, a[mt], false, bf[nt], (short)0, acc[mt][nt], false, false);
  }
  int ln = lane & 15, hb = lane >> 4;
  const float* gb  = grad + b * U;
  const float* db  = diag + b * U;
  const float* m1b = m1v + b * U;
  const float* m2b = m2v + b * U;
#pragma unroll
  for (int nt = 0; nt < 2; ++nt) {
    int l = tn0 + nt * 16 + ln;
    float gl  = gb[l];
    float m1l = m1b[l];
    float m2l = m2b[l];
#pragma unroll
    for (int mt = 0; mt < 4; ++mt)
#pragma unroll
      for (int r = 0; r < 8; ++r) {
        int ii = tm0 + mt * 16 + r + hb * 8;
        float s = acc[mt][nt][r];
        if (ii == l) s += db[ii];
        s *= gb[ii] * gl;
        size_t off = (size_t)b * U * U + (size_t)ii * U + l;
        if (mode == 1) {
          outp[off] = s;
        } else {
          float sv = Sin[off];
          float hm = sv * s + m1b[ii] * m1l * s + m2b[ii] * m2l * sv;
          outp[off] = (mode == 0) ? hm : (outp[off] + hm);
        }
      }
  }
}

// ---------------- final: Sigma_tanhc, Sigma_out, _clean both ------------------
__global__ __launch_bounds__(256) void final_kernel(const float* __restrict__ tcv,
                                                    const float* __restrict__ gtcv,
                                                    const float* __restrict__ muo,
                                                    float* __restrict__ Sout,
                                                    float* __restrict__ Soc) {
  size_t idx = (size_t)blockIdx.x * blockDim.x + threadIdx.x;
  int l = (int)(idx & (UDIM - 1));
  size_t t = idx >> 9;
  int i = (int)(t & (UDIM - 1));
  int b = (int)(t >> 9);
  int bi = b * UDIM + i, bl = b * UDIM + l;
  float soc = Soc[idx];
  float so  = Sout[idx];
  float st  = soc * gtcv[bi] * gtcv[bl];
  float sout = st * so + tcv[bi] * tcv[bl] * so + muo[bi] * muo[bl] * st;
  soc  = (isnan(soc)  || isinf(soc))  ? 0.f : soc;
  sout = (isnan(sout) || isinf(sout)) ? 0.f : sout;
  if (i == l) { soc = fabsf(soc); sout = fabsf(sout); }
  Sout[idx] = sout;
  Soc[idx]  = soc;
}

// -----------------------------------------------------------------------------
extern "C" void kernel_launch(void* const* d_in, const int* in_sizes, int n_in,
                              void* d_out, int out_size, void* d_ws, size_t ws_size,
                              hipStream_t stream) {
  (void)in_sizes; (void)n_in; (void)out_size; (void)ws_size;
  const int B = BDIM, U = UDIM;
  const float* x  = (const float*)d_in[0];
  const float* h  = (const float*)d_in[1];
  const float* ci = (const float*)d_in[2];
  const float* Sh = (const float*)d_in[3];  // Sigma_state
  const float* Sc = (const float*)d_in[4];  // Sigma_istate
  GateParams gp;
  for (int g = 0; g < 4; ++g) {
    gp.U[g]    = (const float*)d_in[5 + 4 * g];
    gp.usig[g] = (const float*)d_in[6 + 4 * g];
    gp.W[g]    = (const float*)d_in[7 + 4 * g];
    gp.wsig[g] = (const float*)d_in[8 + 4 * g];
  }
  // workspace layout
  char* p = (char*)d_ws;
  _Float16* Wt = (_Float16*)p; p += (size_t)4 * U * U * sizeof(_Float16);
  _Float16* St = (_Float16*)p; p += (size_t)B * U * U * sizeof(_Float16);
  _Float16* T  = (_Float16*)p; p += (size_t)B * U * U * sizeof(_Float16);
  float* mu   = (float*)p; p += (size_t)4 * B * U * sizeof(float);
  float* grad = (float*)p; p += (size_t)4 * B * U * sizeof(float);
  float* diag = (float*)p; p += (size_t)4 * B * U * sizeof(float);
  float* tcv  = (float*)p; p += (size_t)B * U * sizeof(float);
  float* gtcv = (float*)p; p += (size_t)B * U * sizeof(float);
  // outputs: (mu_out, c, Sigma_out, Sigma_out_c)
  float* out    = (float*)d_out;
  float* out_mu = out;
  float* out_c  = out + (size_t)B * U;
  float* Sout   = out + (size_t)2 * B * U;
  float* Soc    = Sout + (size_t)B * U * U;

  // 1) transposes to f16 operands
  transpose_f32_f16<<<dim3(16, 16, B), 256, 0, stream>>>(Sh, St, U, U);
  for (int g = 0; g < 4; ++g)
    transpose_f32_f16<<<dim3(16, 16, 1), 256, 0, stream>>>(gp.W[g], Wt + (size_t)g * U * U, U, U);

  // 2) gate means/grads/diag, c, tanh(c), mu_out
  gates_kernel<<<B, U, 0, stream>>>(x, h, ci, Sh, gp, mu, grad, diag, tcv, gtcv, out_mu, out_c);

  // 3) per-gate chained GEMMs; order: f(0), i(1), g(3), o(2)
  const int GS = B * U;
  dim3 gemm_grid(4, 4, B);
  // gate f: Soc = H(Sigma_istate, Sf, prev_istate, f)
  stage1_kernel<<<gemm_grid, 256, 0, stream>>>(Wt + (size_t)0 * U * U, St, T);
  stage2_kernel<<<gemm_grid, 256, 0, stream>>>(T, Wt + (size_t)0 * U * U, grad + 0 * GS,
                                               diag + 0 * GS, Sc, ci, mu + 0 * GS, Soc, 0);
  // gate i: stash Si in Sout region
  stage1_kernel<<<gemm_grid, 256, 0, stream>>>(Wt + (size_t)1 * U * U, St, T);
  stage2_kernel<<<gemm_grid, 256, 0, stream>>>(T, Wt + (size_t)1 * U * U, grad + 1 * GS,
                                               diag + 1 * GS, Sc, ci, mu + 1 * GS, Sout, 1);
  // gate g: Soc += H(Si, Sg, mu_i, mu_g)
  stage1_kernel<<<gemm_grid, 256, 0, stream>>>(Wt + (size_t)3 * U * U, St, T);
  stage2_kernel<<<gemm_grid, 256, 0, stream>>>(T, Wt + (size_t)3 * U * U, grad + 3 * GS,
                                               diag + 3 * GS, Sout, mu + 1 * GS, mu + 3 * GS,
                                               Soc, 2);
  // gate o: stash So in Sout region
  stage1_kernel<<<gemm_grid, 256, 0, stream>>>(Wt + (size_t)2 * U * U, St, T);
  stage2_kernel<<<gemm_grid, 256, 0, stream>>>(T, Wt + (size_t)2 * U * U, grad + 2 * GS,
                                               diag + 2 * GS, Sc, ci, mu + 2 * GS, Sout, 1);

  // 4) Sigma_tanhc / Sigma_out / clean
  final_kernel<<<(B * U * U) / 256, 256, 0, stream>>>(tcv, gtcv, mu + 2 * GS, Sout, Soc);
}
